// RotaryMultiDotProductionAttention_13194139533686
// MI455X (gfx1250) — matrix operand order repaired and checked by reading, not compiled, not run on hardware
//
#include <hip/hip_runtime.h>
#include <hip/hip_bf16.h>
#include <stdint.h>

// ---------------------------------------------------------------------------
// CDNA5 (gfx1250) fused RoPE multi-head attention.
// bf16 WMMA everywhere; projection GEMMs use the Tensor Data Mover (TDM) to
// DMA A/B tiles into LDS (double buffered, TENSORcnt-tracked).
// ---------------------------------------------------------------------------

typedef __attribute__((ext_vector_type(16))) __bf16 v16bf;
typedef __attribute__((ext_vector_type(8)))  float  v8f;
typedef unsigned int u32x4 __attribute__((ext_vector_type(4)));
typedef int          i32x4 __attribute__((ext_vector_type(4)));
typedef int          i32x8 __attribute__((ext_vector_type(8)));

union FragB16 { v16bf v; uint4 q[2]; };
static_assert(sizeof(FragB16) == 32, "frag size");

#define S_LEN   2048
#define D_MODEL 2048
#define N_HEADS 16
#define HEAD_D  128
#define BATCH   2
#define M_ROWS  (BATCH * S_LEN)   // 4096

#if defined(__has_builtin)
#if __has_builtin(__builtin_amdgcn_tensor_load_to_lds) && \
    __has_builtin(__builtin_amdgcn_s_wait_tensorcnt)
#define HAVE_TDM 1
#endif
#endif
#ifndef HAVE_TDM
#define HAVE_TDM 0
#endif

__device__ __forceinline__ unsigned short f2bf(float f) {
    union { float f; uint32_t u; } x; x.f = f;
    uint32_t r = x.u + 0x7FFFu + ((x.u >> 16) & 1u);   // round-to-nearest-even
    return (unsigned short)(r >> 16);
}

// --------------------------- fp32 -> bf16 convert ---------------------------
__global__ void cvt_bf16_kernel(const float* __restrict__ in,
                                unsigned short* __restrict__ out, int n) {
    int i = blockIdx.x * blockDim.x + threadIdx.x;
    if (i < n) out[i] = f2bf(in[i]);
}

#if HAVE_TDM
// --------------------------- TDM 2-D tile -> LDS ----------------------------
// Loads a tile_x x tile_y tile of bf16 (data_size=2B) from a row-major tensor
// [tensor_y][tensor_x] into LDS at lds_off.  pad_enable inserts 16B of LDS
// padding after every 64B row (tile_x=32 -> padded row stride 80B), making the
// subsequent per-lane ds_load_b128 fragment reads bank-conflict-free.
__device__ __forceinline__ void tdm_load_tile_2d(
        const unsigned short* gptr, unsigned lds_off,
        unsigned tensor_x, unsigned tensor_y,
        unsigned tile_x, unsigned tile_y) {
    unsigned long long ga = (unsigned long long)(uintptr_t)gptr;
    u32x4 g0;
    g0[0] = 1u;                                   // count=1 (valid user D#)
    g0[1] = lds_off;                              // lds_addr (bytes)
    g0[2] = (unsigned)ga;                         // global_addr[31:0]
    g0[3] = (unsigned)((ga >> 32) & 0x01FFFFFFu)  // global_addr[56:32]
            | (2u << 30);                         // type=2 ("image")
    i32x8 g1;
    g1[0] = (int)((1u << 16)                      // data_size = 2 bytes
            | (1u << 20)                          // pad_enable
            | (3u << 22)                          // pad_interval: 16 DW = 64B
            | (3u << 25));                        // pad_amount:   4 DW = 16B
    g1[1] = (int)((tensor_x & 0xFFFFu) << 16);    // tensor_dim0[15:0]
    g1[2] = (int)((tensor_x >> 16) | ((tensor_y & 0xFFFFu) << 16));
    g1[3] = (int)((tensor_y >> 16) | (tile_x << 16));     // tile_dim0
    g1[4] = (int)tile_y;                          // tile_dim1 (tile_dim2 = 0)
    g1[5] = (int)tensor_x;                        // tensor_dim0_stride[31:0]
    g1[6] = 0;                                    // stride0[47:32], stride1[15:0]
    g1[7] = 0;                                    // stride1[47:16]
    i32x4 gz = {0, 0, 0, 0};
#if defined(__clang_major__) && (__clang_major__ >= 23)
    i32x8 gz8 = {0, 0, 0, 0, 0, 0, 0, 0};
    __builtin_amdgcn_tensor_load_to_lds(g0, g1, gz, gz, gz8, 0);
#else
    __builtin_amdgcn_tensor_load_to_lds(g0, g1, gz, gz, 0);
#endif
}

// --------------------------- TDM-staged bf16 WMMA GEMM ----------------------
// C = A @ W^T + bias.  A:[M][K] bf16, W:[N][K] bf16, C:[M][N] fp32.
// Block 256 threads (8 waves), block tile 128x128, K-step 32.  Wave 0 drives
// the TDM double buffer; all waves consume LDS fragments with ds_load_b128.
__global__ __launch_bounds__(256) void gemm_bf16_tdm_kernel(
        const unsigned short* __restrict__ A,
        const unsigned short* __restrict__ W,
        const float* __restrict__ bias,
        float* __restrict__ C,
        int M, int N, int K) {
    extern __shared__ char smem[];                // 4 x 10240B tiles
    const int lane  = threadIdx.x & 31;
    const int wave  = threadIdx.x >> 5;
    const int waveM = wave & 3, waveN = wave >> 2;
    const int lr = lane & 15, lh = lane >> 4;
    const int mBlk = blockIdx.x * 128;
    const int nBlk = blockIdx.y * 128;

    const unsigned ROWB   = 80;                   // 64B data + 16B TDM pad
    const unsigned TILE_B = 128 * 80;             // 10240
    // LDS byte offset of smem (flat LDS address low 32 bits == LDS offset).
    const unsigned ldsBase = (unsigned)(uintptr_t)(void*)smem;

    const int nIter = K / 32;
    if (wave == 0) {
        tdm_load_tile_2d(A + (size_t)mBlk * K, ldsBase,
                         (unsigned)K, (unsigned)M, 32, 128);
        tdm_load_tile_2d(W + (size_t)nBlk * K, ldsBase + 2 * TILE_B,
                         (unsigned)K, (unsigned)N, 32, 128);
    }

    v8f acc[2][4];
    #pragma unroll
    for (int i = 0; i < 2; i++)
        #pragma unroll
        for (int j = 0; j < 4; j++) acc[i][j] = {};

    for (int it = 0; it < nIter; ++it) {
        if (wave == 0) {
            if (it + 1 < nIter) {
                const unsigned nb = (unsigned)((it + 1) & 1) * TILE_B;
                tdm_load_tile_2d(A + (size_t)mBlk * K + (it + 1) * 32,
                                 ldsBase + nb,
                                 (unsigned)K, (unsigned)M, 32, 128);
                tdm_load_tile_2d(W + (size_t)nBlk * K + (it + 1) * 32,
                                 ldsBase + 2 * TILE_B + nb,
                                 (unsigned)K, (unsigned)N, 32, 128);
                __builtin_amdgcn_s_wait_tensorcnt(2);  // current buffer landed
            } else {
                __builtin_amdgcn_s_wait_tensorcnt(0);
            }
        }
        __syncthreads();                               // publish buffer it&1

        const unsigned cb = (unsigned)(it & 1) * TILE_B;
        const char* Ab = smem + cb;
        const char* Bb = smem + 2 * TILE_B + cb;
        FragB16 a[2], b[4];
        #pragma unroll
        for (int mi = 0; mi < 2; mi++) {
            // A frag 16x32: row m, K = lh*8+{0..7} then +16
            const char* p = Ab + (unsigned)(waveM * 32 + mi * 16 + lr) * ROWB + lh * 16;
            a[mi].q[0] = *(const uint4*)p;
            a[mi].q[1] = *(const uint4*)(p + 32);
        }
        #pragma unroll
        for (int ni = 0; ni < 4; ni++) {
            // B frag 32x16: col n, K = lh*8+{0..7} then +16
            const char* p = Bb + (unsigned)(waveN * 64 + ni * 16 + lr) * ROWB + lh * 16;
            b[ni].q[0] = *(const uint4*)p;
            b[ni].q[1] = *(const uint4*)(p + 32);
        }
        #pragma unroll
        for (int mi = 0; mi < 2; mi++)
            #pragma unroll
            for (int ni = 0; ni < 4; ni++)
                acc[mi][ni] = __builtin_amdgcn_wmma_f32_16x16x32_bf16(
                    false, a[mi].v, false, b[ni].v, (short)0, acc[mi][ni], false, false);
        __syncthreads();                               // buffer free for reuse
    }

    #pragma unroll
    for (int ni = 0; ni < 4; ni++) {
        const int col = nBlk + waveN * 64 + ni * 16 + lr;
        const float bs = bias[col];
        #pragma unroll
        for (int mi = 0; mi < 2; mi++) {
            #pragma unroll
            for (int r = 0; r < 8; r++) {
                const int row = mBlk + waveM * 32 + mi * 16 + r + lh * 8;
                C[(size_t)row * N + col] = acc[mi][ni][r] + bs;
            }
        }
    }
}
#endif  // HAVE_TDM

// --------------------------- direct-global bf16 WMMA GEMM (fallback) --------
__global__ __launch_bounds__(256) void gemm_bf16_kernel(
        const unsigned short* __restrict__ A,
        const unsigned short* __restrict__ W,
        const float* __restrict__ bias,
        float* __restrict__ C,
        int M, int N, int K) {
    const int lane  = threadIdx.x & 31;
    const int wave  = threadIdx.x >> 5;
    const int waveM = wave & 3;
    const int waveN = wave >> 2;
    const int m0 = blockIdx.x * 128 + waveM * 32;
    const int n0 = blockIdx.y * 128 + waveN * 64;
    const int lr = lane & 15;
    const int lh = lane >> 4;

    v8f acc[2][4];
    #pragma unroll
    for (int i = 0; i < 2; i++)
        #pragma unroll
        for (int j = 0; j < 4; j++) acc[i][j] = {};

    for (int k0 = 0; k0 < K; k0 += 32) {
        FragB16 a[2], b[4];
        #pragma unroll
        for (int mi = 0; mi < 2; mi++) {
            const unsigned short* p = A + (size_t)(m0 + mi * 16 + lr) * K + k0 + lh * 8;
            a[mi].q[0] = *(const uint4*)p;
            a[mi].q[1] = *(const uint4*)(p + 16);
        }
        #pragma unroll
        for (int ni = 0; ni < 4; ni++) {
            const unsigned short* p = W + (size_t)(n0 + ni * 16 + lr) * K + k0 + lh * 8;
            b[ni].q[0] = *(const uint4*)p;
            b[ni].q[1] = *(const uint4*)(p + 16);
        }
        #pragma unroll
        for (int mi = 0; mi < 2; mi++)
            #pragma unroll
            for (int ni = 0; ni < 4; ni++)
                acc[mi][ni] = __builtin_amdgcn_wmma_f32_16x16x32_bf16(
                    false, a[mi].v, false, b[ni].v, (short)0, acc[mi][ni], false, false);
    }

    #pragma unroll
    for (int ni = 0; ni < 4; ni++) {
        const int col = n0 + ni * 16 + lr;
        const float bs = bias[col];
        #pragma unroll
        for (int mi = 0; mi < 2; mi++) {
            #pragma unroll
            for (int r = 0; r < 8; r++) {
                const int row = m0 + mi * 16 + r + lh * 8;
                C[(size_t)row * N + col] = acc[mi][ni][r] + bs;
            }
        }
    }
}

// --------------------------- RoPE + head split (fp32 -> bf16) ---------------
__global__ void rope_split_kernel(const float* __restrict__ src,
                                  unsigned short* __restrict__ dst) {
    int idx = blockIdx.x * blockDim.x + threadIdx.x;   // over b*NH*S*(HEAD_D/2)
    int i   = idx & 63;
    int t   = idx >> 6;
    int pos = t & (S_LEN - 1);
    int bn  = t >> 11;
    int head = bn & (N_HEADS - 1);
    int bb   = bn >> 4;
    float inv = __expf(-(float)(2 * i) * (9.210340371976184f / 128.0f));
    float ang = (float)pos * inv;
    float s, c;
    __sincosf(ang, &s, &c);
    const float* sp = src + ((size_t)(bb * S_LEN + pos)) * D_MODEL + head * HEAD_D + 2 * i;
    float x0 = sp[0], x1 = sp[1];
    unsigned short* dp = dst + ((size_t)bn * S_LEN + pos) * HEAD_D + 2 * i;
    dp[0] = f2bf(x0 * c - x1 * s);
    dp[1] = f2bf(x1 * c + x0 * s);
}

// --------------------------- V transpose (fp32 -> bf16) ---------------------
__global__ void vtrans_kernel(const float* __restrict__ src,
                              unsigned short* __restrict__ dst) {
    int idx = blockIdx.x * blockDim.x + threadIdx.x;   // ((bn*HEAD_D+h)*S+pos)
    int pos = idx & (S_LEN - 1);
    int t   = idx >> 11;
    int h   = t & (HEAD_D - 1);
    int bn  = t >> 7;
    int head = bn & (N_HEADS - 1);
    int bb   = bn >> 4;
    dst[idx] = f2bf(src[((size_t)(bb * S_LEN + pos)) * D_MODEL + head * HEAD_D + h]);
}

// --------------------------- flash attention (bf16 WMMA) --------------------
__global__ __launch_bounds__(128) void attn_kernel(
        const unsigned short* __restrict__ Qh,
        const unsigned short* __restrict__ Kh,
        const unsigned short* __restrict__ Vt,
        const unsigned char*  __restrict__ mask,
        unsigned short* __restrict__ Ctx) {
    __shared__ __align__(16) unsigned short Pl[4][16 * 32];

    const int lane = threadIdx.x & 31;
    const int wave = threadIdx.x >> 5;
    const int lr = lane & 15;
    const int lh = lane >> 4;

    const int hg   = blockIdx.x >> 5;
    const int bb   = hg >> 4;
    const int head = hg & (N_HEADS - 1);
    const int s0   = ((blockIdx.x & 31) * 4 + wave) * 16;

    const unsigned short* Qbase = Qh + (size_t)hg * S_LEN * HEAD_D;
    const unsigned short* Kbase = Kh + (size_t)hg * S_LEN * HEAD_D;
    const unsigned short* Vbase = Vt + (size_t)hg * HEAD_D * S_LEN;
    const unsigned char*  mrow  = mask + (size_t)bb * S_LEN;

    FragB16 qf[4];
    #pragma unroll
    for (int kk = 0; kk < 4; kk++) {
        const unsigned short* p = Qbase + (size_t)(s0 + lr) * HEAD_D + kk * 32 + lh * 8;
        qf[kk].q[0] = *(const uint4*)p;
        qf[kk].q[1] = *(const uint4*)(p + 16);
    }

    v8f O[8];
    #pragma unroll
    for (int t = 0; t < 8; t++) O[t] = {};
    float mrun[8], lrun[8];
    #pragma unroll
    for (int r = 0; r < 8; r++) { mrun[r] = -3.0e38f; lrun[r] = 0.0f; }

    const float scale = 0.08838834764831845f;      // 1/sqrt(128)
    unsigned short* myP = &Pl[wave][0];

    for (int pb = 0; pb < S_LEN; pb += 32) {
        v8f sfr[2];
        #pragma unroll
        for (int t = 0; t < 2; t++) {
            v8f accS = {};
            #pragma unroll
            for (int kk = 0; kk < 4; kk++) {
                FragB16 bk;
                const unsigned short* p =
                    Kbase + (size_t)(pb + t * 16 + lr) * HEAD_D + kk * 32 + lh * 8;
                bk.q[0] = *(const uint4*)p;
                bk.q[1] = *(const uint4*)(p + 16);
                accS = __builtin_amdgcn_wmma_f32_16x16x32_bf16(
                    false, qf[kk].v, false, bk.v, (short)0, accS, false, false);
            }
            sfr[t] = accS;
        }

        const bool keep0 = mrow[pb + lr] != 0;
        const bool keep1 = mrow[pb + 16 + lr] != 0;
        #pragma unroll
        for (int r = 0; r < 8; r++) {
            sfr[0][r] = keep0 ? sfr[0][r] * scale : -1.0e8f;
            sfr[1][r] = keep1 ? sfr[1][r] * scale : -1.0e8f;
        }

        float alpha[8];
        #pragma unroll
        for (int r = 0; r < 8; r++) {
            float rmax = fmaxf(sfr[0][r], sfr[1][r]);
            #pragma unroll
            for (int off = 1; off < 16; off <<= 1)
                rmax = fmaxf(rmax, __shfl_xor(rmax, off, 32));
            const float mnew = fmaxf(mrun[r], rmax);
            const float a    = __expf(mrun[r] - mnew);
            const float e0   = __expf(sfr[0][r] - mnew);
            const float e1   = __expf(sfr[1][r] - mnew);
            float rsum = e0 + e1;
            #pragma unroll
            for (int off = 1; off < 16; off <<= 1)
                rsum += __shfl_xor(rsum, off, 32);
            lrun[r] = lrun[r] * a + rsum;
            mrun[r] = mnew;
            alpha[r] = a;
            sfr[0][r] = e0;
            sfr[1][r] = e1;
        }
        #pragma unroll
        for (int t = 0; t < 8; t++)
            #pragma unroll
            for (int r = 0; r < 8; r++) O[t][r] *= alpha[r];

        #pragma unroll
        for (int r = 0; r < 8; r++) {
            const int row = r + lh * 8;
            myP[row * 32 + lr]      = f2bf(sfr[0][r]);
            myP[row * 32 + 16 + lr] = f2bf(sfr[1][r]);
        }
        asm volatile("s_wait_dscnt 0" ::: "memory");

        FragB16 pf;
        {
            const unsigned short* pp = myP + lr * 32 + lh * 8;
            pf.q[0] = *(const uint4*)pp;
            pf.q[1] = *(const uint4*)(pp + 16);
        }

        #pragma unroll
        for (int t = 0; t < 8; t++) {
            FragB16 bv;
            const unsigned short* p =
                Vbase + (size_t)(t * 16 + lr) * S_LEN + pb + lh * 8;
            bv.q[0] = *(const uint4*)p;
            bv.q[1] = *(const uint4*)(p + 16);
            O[t] = __builtin_amdgcn_wmma_f32_16x16x32_bf16(
                false, pf.v, false, bv.v, (short)0, O[t], false, false);
        }
    }

    float rinv[8];
    #pragma unroll
    for (int r = 0; r < 8; r++) rinv[r] = 1.0f / lrun[r];
    #pragma unroll
    for (int t = 0; t < 8; t++) {
        #pragma unroll
        for (int r = 0; r < 8; r++) {
            const int row = s0 + r + lh * 8;
            const int col = head * HEAD_D + t * 16 + lr;
            Ctx[((size_t)bb * S_LEN + row) * D_MODEL + col] = f2bf(O[t][r] * rinv[r]);
        }
    }
}

// ---------------------------------------------------------------------------
extern "C" void kernel_launch(void* const* d_in, const int* in_sizes, int n_in,
                              void* d_out, int out_size, void* d_ws, size_t ws_size,
                              hipStream_t stream) {
    (void)in_sizes; (void)n_in; (void)out_size; (void)ws_size;

    const float* q  = (const float*)d_in[0];
    const float* k  = (const float*)d_in[1];
    const float* v  = (const float*)d_in[2];
    const unsigned char* amask = (const unsigned char*)d_in[3];
    const float* Wq = (const float*)d_in[4];
    const float* bq = (const float*)d_in[5];
    const float* Wk = (const float*)d_in[6];
    const float* bk = (const float*)d_in[7];
    const float* Wv = (const float*)d_in[8];
    const float* bv = (const float*)d_in[9];
    const float* Wo = (const float*)d_in[10];
    const float* bo = (const float*)d_in[11];
    float* out = (float*)d_out;

    char* ws = (char*)d_ws;
    const size_t ACT_B = (size_t)M_ROWS * D_MODEL * 2;   // 16 MiB bf16 activations
    const size_t W_B   = (size_t)D_MODEL * D_MODEL * 2;  //  8 MiB bf16 weights
    const size_t PRJ_B = (size_t)M_ROWS * D_MODEL * 4;   // 32 MiB fp32 projections

    unsigned short* Xq  = (unsigned short*)(ws);                    // later Qh
    unsigned short* Xk  = (unsigned short*)(ws + ACT_B);            // later Kh
    unsigned short* Xv  = (unsigned short*)(ws + 2 * ACT_B);        // later Vt
    unsigned short* Wqb = (unsigned short*)(ws + 3 * ACT_B);
    unsigned short* Wkb = (unsigned short*)(ws + 3 * ACT_B + W_B);
    unsigned short* Wvb = (unsigned short*)(ws + 3 * ACT_B + 2 * W_B);
    unsigned short* Wob = (unsigned short*)(ws + 3 * ACT_B + 3 * W_B);
    float* Pq = (float*)(ws + 3 * ACT_B + 4 * W_B);                 // later Ctx
    float* Pk = (float*)(ws + 3 * ACT_B + 4 * W_B + PRJ_B);
    float* Pv = (float*)(ws + 3 * ACT_B + 4 * W_B + 2 * PRJ_B);
    unsigned short* Qh  = Xq;               // alias: Xq dead after gemm q
    unsigned short* Kh  = Xk;
    unsigned short* Vt  = Xv;
    unsigned short* Ctx = (unsigned short*)Pq;   // alias: Pq dead after rope q

    const int nAct = M_ROWS * D_MODEL;       // 8388608
    const int nW   = D_MODEL * D_MODEL;      // 4194304

    // 1) bf16 conversions
    cvt_bf16_kernel<<<(nAct + 255) / 256, 256, 0, stream>>>(q, Xq, nAct);
    cvt_bf16_kernel<<<(nAct + 255) / 256, 256, 0, stream>>>(k, Xk, nAct);
    cvt_bf16_kernel<<<(nAct + 255) / 256, 256, 0, stream>>>(v, Xv, nAct);
    cvt_bf16_kernel<<<(nW + 255) / 256, 256, 0, stream>>>(Wq, Wqb, nW);
    cvt_bf16_kernel<<<(nW + 255) / 256, 256, 0, stream>>>(Wk, Wkb, nW);
    cvt_bf16_kernel<<<(nW + 255) / 256, 256, 0, stream>>>(Wv, Wvb, nW);
    cvt_bf16_kernel<<<(nW + 255) / 256, 256, 0, stream>>>(Wo, Wob, nW);

    // 2) projections (TDM-staged WMMA when available)
    dim3 ggrid(M_ROWS / 128, D_MODEL / 128);
#if HAVE_TDM
    const unsigned LDS_BYTES = 4u * 128u * 80u;  // 40 KiB double-buffered tiles
    gemm_bf16_tdm_kernel<<<ggrid, 256, LDS_BYTES, stream>>>(Xq, Wqb, bq, Pq, M_ROWS, D_MODEL, D_MODEL);
    gemm_bf16_tdm_kernel<<<ggrid, 256, LDS_BYTES, stream>>>(Xk, Wkb, bk, Pk, M_ROWS, D_MODEL, D_MODEL);
    gemm_bf16_tdm_kernel<<<ggrid, 256, LDS_BYTES, stream>>>(Xv, Wvb, bv, Pv, M_ROWS, D_MODEL, D_MODEL);
#else
    gemm_bf16_kernel<<<ggrid, 256, 0, stream>>>(Xq, Wqb, bq, Pq, M_ROWS, D_MODEL, D_MODEL);
    gemm_bf16_kernel<<<ggrid, 256, 0, stream>>>(Xk, Wkb, bk, Pk, M_ROWS, D_MODEL, D_MODEL);
    gemm_bf16_kernel<<<ggrid, 256, 0, stream>>>(Xv, Wvb, bv, Pv, M_ROWS, D_MODEL, D_MODEL);
#endif

    // 3) RoPE + head split; V transpose
    const int nRope = BATCH * N_HEADS * S_LEN * (HEAD_D / 2);   // 4194304
    rope_split_kernel<<<nRope / 256, 256, 0, stream>>>(Pq, Qh);
    rope_split_kernel<<<nRope / 256, 256, 0, stream>>>(Pk, Kh);
    vtrans_kernel<<<nAct / 256, 256, 0, stream>>>(Pv, Vt);

    // 4) flash attention (WMMA)
    attn_kernel<<<BATCH * N_HEADS * 32, 128, 0, stream>>>(Qh, Kh, Vt, amask, Ctx);

    // 5) output projection
#if HAVE_TDM
    gemm_bf16_tdm_kernel<<<ggrid, 256, 4u * 128u * 80u, stream>>>(Ctx, Wob, bo, out, M_ROWS, D_MODEL, D_MODEL);
#else
    gemm_bf16_kernel<<<ggrid, 256, 0, stream>>>(Ctx, Wob, bo, out, M_ROWS, D_MODEL, D_MODEL);
#endif
}
